// RNNDecoder_46239617909508
// MI455X (gfx1250) — compile-verified, matrix operand order checked by reading, not run on hardware
//
#include <hip/hip_runtime.h>
#include <hip/hip_bf16.h>
#include <math.h>

// Problem dims (from reference): S,B,E,L,T,V
#define S_ 256
#define B_ 128
#define E_ 1024
#define L_ 2
#define T_ 32
#define V_ 32000
#define LN_EPS_ 1e-5f

// Flip to 0 if the assembler rejects the async-to-LDS mnemonic.
#define GFX1250_ASYNC_LDS 1

typedef __attribute__((ext_vector_type(16))) __bf16 v16bf;
typedef __attribute__((ext_vector_type(8)))  float  v8f;

// ---------------- bf16 helpers (memory type = unsigned short) ----------------
__device__ __forceinline__ float bf2f(unsigned short u) {
  union { unsigned int i; float f; } c; c.i = ((unsigned int)u) << 16; return c.f;
}
__device__ __forceinline__ unsigned short f2bf(float f) {
  union { float f; unsigned int i; } c; c.f = f;
  unsigned int r = c.i + 0x7FFFu + ((c.i >> 16) & 1u);  // round-to-nearest-even
  return (unsigned short)(r >> 16);
}

// ---------------- async global->LDS copy (gfx1250, ASYNCcnt-tracked) ----------------
// global_load_async_to_lds_b128: VDST = VGPR with LDS byte offset, VADDR = 64-bit
// global address (CDNA5 ISA opcode 98).
__device__ __forceinline__ void async_copy_b128(unsigned lds_byte_off, const void* gaddr) {
#if GFX1250_ASYNC_LDS
  asm volatile("global_load_async_to_lds_b128 %0, %1, off"
               :
               : "v"(lds_byte_off), "v"((unsigned long long)(size_t)gaddr)
               : "memory");
#else
  (void)lds_byte_off; (void)gaddr;
#endif
}
__device__ __forceinline__ void wait_async_zero() {
#if GFX1250_ASYNC_LDS
  asm volatile("s_wait_asynccnt 0" ::: "memory");
#endif
}

// ---------------- WMMA fragment load/store (wave32, 16x16 tiles) ----------------
// A fragment 16x32 bf16 (row-major, K inner): lane = 16*khalf + m;
//   lane<16 holds K=0..7 & 16..23, lane>=16 holds K=8..15 & 24..31 (2 x 16B chunks).
// B fragment 32x16 bf16 for NT GEMM (W stored NxK row-major): identical pattern with
//   rows indexed by N, so one loader serves both operands.
union FragU { v16bf v; uint4 q[2]; };

__device__ __forceinline__ v16bf load_frag_g(const unsigned short* __restrict__ p,
                                             int ld, int row0, int k0) {
  int lane = threadIdx.x & 31;
  const unsigned short* base =
      p + (size_t)(row0 + (lane & 15)) * ld + (k0 + ((lane >> 4) << 3));
  FragU f;
  f.q[0] = *(const uint4*)(base);
  f.q[1] = *(const uint4*)(base + 16);
  return f.v;
}

// A fragment from the LDS slab (row stride = 32 bf16 = 64B) -> ds_load_b128 x2
__device__ __forceinline__ v16bf load_afrag_lds(const unsigned short* slab, int row0) {
  int lane = threadIdx.x & 31;
  const unsigned short* base = slab + (row0 + (lane & 15)) * 32 + ((lane >> 4) << 3);
  FragU f;
  f.q[0] = *(const uint4*)(base);
  f.q[1] = *(const uint4*)(base + 16);
  return f.v;
}

// C/D 16x16 f32 fragment: lane = 16*mhalf + n, VGPR r holds row m = 8*mhalf + r.
__device__ __forceinline__ v8f load_cfrag(const float* __restrict__ C, int ldc,
                                          int row0, int col0) {
  int lane = threadIdx.x & 31;
  const float* p = C + (size_t)(row0 + ((lane >> 4) << 3)) * ldc + col0 + (lane & 15);
  v8f c;
#pragma unroll
  for (int r = 0; r < 8; ++r) c[r] = p[(size_t)r * ldc];
  return c;
}
__device__ __forceinline__ void store_cfrag(float* __restrict__ C, int ldc,
                                            int row0, int col0, v8f c) {
  int lane = threadIdx.x & 31;
  float* p = C + (size_t)(row0 + ((lane >> 4) << 3)) * ldc + col0 + (lane & 15);
#pragma unroll
  for (int r = 0; r < 8; ++r) p[(size_t)r * ldc] = c[r];
}

#define WMMA_BF16(a, b, c) \
  __builtin_amdgcn_wmma_f32_16x16x32_bf16(false, (a), false, (b), (short)0, (c), false, false)

// ---------------- NT GEMM: C(128xN,f32) (+)= A(128xK,bf16) * W(NxK,bf16)^T ----------------
// Block = 256 threads = 8 waves, block tile = 128(M) x 128(N).
// Wave tile = 32x64 (2x4 WMMA tiles): A-fragments reused 4x, B-fragments 2x.
// A K-slab (128x32 bf16 = 8KB) is staged in LDS with double-buffered async copies
// shared by all 8 waves; B is streamed from global (L2-resident across steps).
// Requires: M==128, N%128==0, K%32==0. No early exits -> EXEC all-1s for WMMA.
__global__ void __launch_bounds__(256)
k_gemm_bf16_nt(const unsigned short* __restrict__ A, int lda,
               const unsigned short* __restrict__ W, int ldb,
               float* __restrict__ C, int ldc,
               int N, int K, int accumulate) {
  __shared__ unsigned short Asl[2][128 * 32];  // 2 x 8KB

  const int tid  = threadIdx.x;
  const int lane = tid & 31;
  const int wid  = tid >> 5;
  const int m0   = (wid & 3) << 5;                        // 0,32,64,96
  const int n0   = blockIdx.x * 128 + ((wid >> 2) << 6);  // +0 or +64

  // Cooperative A-slab stage: each thread covers 32B (=16 bf16) of the 8KB slab.
  const int arow = tid >> 1;            // 0..127
  const int acol = (tid & 1) << 4;      // element 0 or 16
  const unsigned lds_lin = (unsigned)tid * 32;  // byte offset within a slab buffer
  const unsigned lds_base = (unsigned)(size_t)(&Asl[0][0]);

  auto stage_A = [&](int kb, int buf) {
    const unsigned short* src = A + (size_t)arow * lda + (kb << 5) + acol;
    unsigned dst = lds_base + (unsigned)buf * (128 * 32 * 2) + lds_lin;
#if GFX1250_ASYNC_LDS
    async_copy_b128(dst, src);
    async_copy_b128(dst + 16, src + 8);
#else
    uint4 q0 = *(const uint4*)src;
    uint4 q1 = *(const uint4*)(src + 8);
    unsigned short* d = &Asl[buf][0] + (size_t)tid * 16;
    *(uint4*)(d) = q0;
    *(uint4*)(d + 8) = q1;
#endif
  };

  v8f acc[2][4];
  if (accumulate) {
#pragma unroll
    for (int i = 0; i < 2; ++i)
#pragma unroll
      for (int j = 0; j < 4; ++j)
        acc[i][j] = load_cfrag(C, ldc, m0 + 16 * i, n0 + 16 * j);
  } else {
#pragma unroll
    for (int i = 0; i < 2; ++i)
#pragma unroll
      for (int j = 0; j < 4; ++j)
        acc[i][j] = v8f{};
  }

  const int nk = K >> 5;
  stage_A(0, 0);
  wait_async_zero();
  __syncthreads();

  for (int kb = 0; kb < nk; ++kb) {
    const int buf = kb & 1;
    if (kb + 1 < nk) {
      stage_A(kb + 1, buf ^ 1);  // async DMA overlaps this slab's WMMAs
      // prefetch next K-slab of the streamed W rows (64 rows per wave)
      const unsigned short* pf0 = W + (size_t)(n0 + lane) * ldb + (kb << 5) + 32;
      const unsigned short* pf1 = W + (size_t)(n0 + 32 + lane) * ldb + (kb << 5) + 32;
      __builtin_prefetch((const void*)pf0, 0, 1);
      __builtin_prefetch((const void*)pf1, 0, 1);
    }
    const int k = kb << 5;
    v16bf a0 = load_afrag_lds(Asl[buf], m0);
    v16bf a1 = load_afrag_lds(Asl[buf], m0 + 16);
#pragma unroll
    for (int j = 0; j < 4; ++j) {
      v16bf bj = load_frag_g(W, ldb, n0 + 16 * j, k);
      acc[0][j] = WMMA_BF16(a0, bj, acc[0][j]);
      acc[1][j] = WMMA_BF16(a1, bj, acc[1][j]);
    }
    wait_async_zero();
    __syncthreads();
  }

#pragma unroll
  for (int i = 0; i < 2; ++i)
#pragma unroll
    for (int j = 0; j < 4; ++j)
      store_cfrag(C, ldc, m0 + 16 * i, n0 + 16 * j, acc[i][j]);
}

// ---------------- utility / pointwise kernels ----------------
__global__ void __launch_bounds__(256)
k_f32_to_bf16(const float* __restrict__ src, unsigned short* __restrict__ dst, long n) {
  long i = (long)blockIdx.x * blockDim.x + threadIdx.x;
  long stride = (long)gridDim.x * blockDim.x;
  for (; i < n; i += stride) dst[i] = f2bf(src[i]);
}

// feats (S,B,E) f32 -> (B,S,E) bf16 for coalesced attention access
__global__ void __launch_bounds__(256)
k_feats_bt(const float* __restrict__ feats, unsigned short* __restrict__ out) {
  long idx = (long)blockIdx.x * blockDim.x + threadIdx.x;  // over B*S*E
  long e  = idx & (E_ - 1);
  long bs = idx >> 10;
  long s  = bs & (S_ - 1);
  long b  = bs >> 8;
  out[idx] = f2bf(feats[(s * B_ + b) * E_ + e]);
}

// h0 = c0 = mean_s feats; broadcast to both layers, keep c in f32, h in bf16
__global__ void __launch_bounds__(256)
k_init_state(const float* __restrict__ feats, float* __restrict__ c,
             unsigned short* __restrict__ hbf) {
  int idx = blockIdx.x * blockDim.x + threadIdx.x;  // b*E + e
  const float* p = feats + idx;
  float acc = 0.f;
  for (int s = 0; s < S_; ++s) acc += p[(size_t)s * B_ * E_];
  float m = acc * (1.f / S_);
  unsigned short mb = f2bf(m);
  c[idx] = m;    c[B_ * E_ + idx] = m;
  hbf[idx] = mb; hbf[B_ * E_ + idx] = mb;
}

// gates(B,4E) f32 + biases -> i,f,g,o; update c (f32) and h (bf16, feeds next GEMMs)
__global__ void __launch_bounds__(256)
k_lstm_elem(const float* __restrict__ gates,
            const float* __restrict__ bih, const float* __restrict__ bhh,
            float* __restrict__ c, unsigned short* __restrict__ h_bf) {
  int idx = blockIdx.x * blockDim.x + threadIdx.x;  // b*E + e
  int b = idx >> 10, e = idx & (E_ - 1);
  const float* g = gates + (size_t)b * (4 * E_);
  float gi = g[e]          + bih[e]          + bhh[e];
  float gf = g[E_ + e]     + bih[E_ + e]     + bhh[E_ + e];
  float gg = g[2 * E_ + e] + bih[2 * E_ + e] + bhh[2 * E_ + e];
  float go = g[3 * E_ + e] + bih[3 * E_ + e] + bhh[3 * E_ + e];
  float si = 1.f / (1.f + __expf(-gi));
  float sf = 1.f / (1.f + __expf(-gf));
  float so = 1.f / (1.f + __expf(-go));
  float cn = sf * c[idx] + si * tanhf(gg);
  float hn = so * tanhf(cn);
  c[idx]    = cn;
  h_bf[idx] = f2bf(hn);
}

// score[b,s] = hq[b,:] . feats_bt[b,s,:] + mask[b,s]   (one wave per (b,s), b128 loads)
__global__ void __launch_bounds__(256)
k_attn_score(const float* __restrict__ hq, const unsigned short* __restrict__ fbt,
             const float* __restrict__ mask, float* __restrict__ score) {
  int gw   = blockIdx.x * (blockDim.x >> 5) + (threadIdx.x >> 5);
  int lane = threadIdx.x & 31;
  int b = gw >> 8;          // S_ == 256
  int s = gw & (S_ - 1);
  const unsigned short* fp = fbt + ((size_t)b * S_ + s) * E_;
  const float* hp = hq + (size_t)b * E_;
  float acc = 0.f;
  for (int e0 = lane * 8; e0 < E_; e0 += 256) {
    uint4 q = *(const uint4*)(fp + e0);
    const float* h8 = hp + e0;
    unsigned int uu[4] = {q.x, q.y, q.z, q.w};
#pragma unroll
    for (int i = 0; i < 4; ++i) {
      acc += h8[2 * i]     * __uint_as_float(uu[i] << 16);
      acc += h8[2 * i + 1] * __uint_as_float(uu[i] & 0xFFFF0000u);
    }
  }
#pragma unroll
  for (int o = 16; o > 0; o >>= 1) acc += __shfl_xor(acc, o, 32);
  if (lane == 0) score[(size_t)b * S_ + s] = acc + mask[(size_t)b * S_ + s];
}

// in-place softmax over S per batch row (block = S_ threads)
__global__ void __launch_bounds__(S_)
k_softmax(float* __restrict__ score) {
  __shared__ float red[S_];
  int b = blockIdx.x, s = threadIdx.x;
  float v = score[(size_t)b * S_ + s];
  red[s] = v; __syncthreads();
  for (int o = S_ / 2; o > 0; o >>= 1) {
    if (s < o) red[s] = fmaxf(red[s], red[s + o]);
    __syncthreads();
  }
  float mx = red[0]; __syncthreads();
  float e = __expf(v - mx);
  red[s] = e; __syncthreads();
  for (int o = S_ / 2; o > 0; o >>= 1) {
    if (s < o) red[s] += red[s + o];
    __syncthreads();
  }
  score[(size_t)b * S_ + s] = e / red[0];
}

// ctx[b,e] = sum_s w[b,s] * feats_bt[b,s,e] ; emit bf16 for the concat GEMM
__global__ void __launch_bounds__(256)
k_ctx(const float* __restrict__ w, const unsigned short* __restrict__ fbt,
      unsigned short* __restrict__ ctx_bf) {
  int idx = blockIdx.x * blockDim.x + threadIdx.x;  // b*E + e
  int b = idx >> 10, e = idx & (E_ - 1);
  const unsigned short* fp = fbt + (size_t)b * S_ * E_ + e;
  const float* wp = w + (size_t)b * S_;
  float acc = 0.f;
  for (int s = 0; s < S_; ++s) acc += wp[s] * bf2f(fp[(size_t)s * E_]);
  ctx_bf[idx] = f2bf(acc);
}

// layernorm + tanh over E per row; emit bf16 A operand for the projection GEMM
__global__ void __launch_bounds__(256)
k_ln_tanh(const float* __restrict__ x, const float* __restrict__ g,
          const float* __restrict__ be, unsigned short* __restrict__ out) {
  __shared__ float red[256];
  int b = blockIdx.x, t = threadIdx.x;
  const float* xp = x + (size_t)b * E_;
  float vals[4]; float sum = 0.f;
#pragma unroll
  for (int i = 0; i < 4; ++i) { vals[i] = xp[t + 256 * i]; sum += vals[i]; }
  red[t] = sum; __syncthreads();
  for (int o = 128; o > 0; o >>= 1) { if (t < o) red[t] += red[t + o]; __syncthreads(); }
  float mu = red[0] * (1.f / E_); __syncthreads();
  float vs = 0.f;
#pragma unroll
  for (int i = 0; i < 4; ++i) { float d = vals[i] - mu; vs += d * d; }
  red[t] = vs; __syncthreads();
  for (int o = 128; o > 0; o >>= 1) { if (t < o) red[t] += red[t + o]; __syncthreads(); }
  float inv = rsqrtf(red[0] * (1.f / E_) + LN_EPS_);
#pragma unroll
  for (int i = 0; i < 4; ++i) {
    int e = t + 256 * i;
    out[(size_t)b * E_ + e] = f2bf(tanhf((vals[i] - mu) * inv * g[e] + be[e]));
  }
}

// ---------------- host orchestration ----------------
extern "C" void kernel_launch(void* const* d_in, const int* in_sizes, int n_in,
                              void* d_out, int out_size, void* d_ws, size_t ws_size,
                              hipStream_t stream) {
  (void)in_sizes; (void)n_in; (void)out_size; (void)ws_size;
  const float* feats    = (const float*)d_in[0];
  const float* mask     = (const float*)d_in[1];
  const float* emb      = (const float*)d_in[2];
  const float* w_ih     = (const float*)d_in[3];
  const float* w_hh     = (const float*)d_in[4];
  const float* b_ih     = (const float*)d_in[5];
  const float* b_hh     = (const float*)d_in[6];
  const float* attn_w   = (const float*)d_in[7];
  const float* concat_w = (const float*)d_in[8];
  const float* ln_g     = (const float*)d_in[9];
  const float* ln_b     = (const float*)d_in[10];
  const float* proj_w   = (const float*)d_in[11];
  float* out = (float*)d_out;

  char* ws = (char*)d_ws;
  size_t off = 0;
  auto alloc = [&](size_t bytes) -> void* {
    void* p = ws + off;
    off = (off + bytes + 255) & ~(size_t)255;
    return p;
  };
  unsigned short* wih_bf  = (unsigned short*)alloc((size_t)L_ * 4 * E_ * E_ * 2);
  unsigned short* whh_bf  = (unsigned short*)alloc((size_t)L_ * 4 * E_ * E_ * 2);
  unsigned short* attn_bf = (unsigned short*)alloc((size_t)E_ * E_ * 2);
  unsigned short* conc_bf = (unsigned short*)alloc((size_t)E_ * 2 * E_ * 2);
  unsigned short* proj_bf = (unsigned short*)alloc((size_t)V_ * E_ * 2);
  unsigned short* fbt     = (unsigned short*)alloc((size_t)B_ * S_ * E_ * 2);
  float*          cst     = (float*)alloc((size_t)L_ * B_ * E_ * 4);
  unsigned short* hbf     = (unsigned short*)alloc((size_t)L_ * B_ * E_ * 2);
  unsigned short* x0bf    = (unsigned short*)alloc((size_t)B_ * E_ * 2);
  float*          gates   = (float*)alloc((size_t)B_ * 4 * E_ * 4);
  float*          hq      = (float*)alloc((size_t)B_ * E_ * 4);
  float*          score   = (float*)alloc((size_t)B_ * S_ * 4);
  unsigned short* ctxbf   = (unsigned short*)alloc((size_t)B_ * E_ * 2);
  float*          cat     = (float*)alloc((size_t)B_ * E_ * 4);
  unsigned short* catbf   = (unsigned short*)alloc((size_t)B_ * E_ * 2);

  // M is always 128; block tile = 128x128 -> grid = N/128
  auto gemm = [&](const unsigned short* A, int lda, const unsigned short* W, int ldb,
                  float* C, int ldc, int N, int K, int acc) {
    k_gemm_bf16_nt<<<N / 128, 256, 0, stream>>>(A, lda, W, ldb, C, ldc, N, K, acc);
  };

  // ---- one-time conversions (re-run each call for determinism) ----
  k_f32_to_bf16<<<4096, 256, 0, stream>>>(w_ih, wih_bf, (long)L_ * 4 * E_ * E_);
  k_f32_to_bf16<<<4096, 256, 0, stream>>>(w_hh, whh_bf, (long)L_ * 4 * E_ * E_);
  k_f32_to_bf16<<<1024, 256, 0, stream>>>(attn_w, attn_bf, (long)E_ * E_);
  k_f32_to_bf16<<<2048, 256, 0, stream>>>(concat_w, conc_bf, (long)E_ * 2 * E_);
  k_f32_to_bf16<<<8192, 256, 0, stream>>>(proj_w, proj_bf, (long)V_ * E_);
  k_feats_bt<<<(B_ * S_ * E_) / 256, 256, 0, stream>>>(feats, fbt);
  k_init_state<<<(B_ * E_) / 256, 256, 0, stream>>>(feats, cst, hbf);

  // ---- T sequential decode steps ----
  for (int t = 0; t < T_; ++t) {
    k_f32_to_bf16<<<512, 256, 0, stream>>>(emb + (size_t)t * B_ * E_, x0bf, (long)B_ * E_);
    const unsigned short* x = x0bf;
    for (int l = 0; l < L_; ++l) {
      gemm(x, E_, wih_bf + (size_t)l * 4 * E_ * E_, E_, gates, 4 * E_, 4 * E_, E_, 0);
      gemm(hbf + (size_t)l * B_ * E_, E_, whh_bf + (size_t)l * 4 * E_ * E_, E_,
           gates, 4 * E_, 4 * E_, E_, 1);
      k_lstm_elem<<<(B_ * E_) / 256, 256, 0, stream>>>(
          gates, b_ih + (size_t)l * 4 * E_, b_hh + (size_t)l * 4 * E_,
          cst + (size_t)l * B_ * E_, hbf + (size_t)l * B_ * E_);
      x = hbf + (size_t)l * B_ * E_;
    }
    // attention
    gemm(hbf + (size_t)(L_ - 1) * B_ * E_, E_, attn_bf, E_, hq, E_, E_, E_, 0);
    k_attn_score<<<(B_ * S_) / 8, 256, 0, stream>>>(hq, fbt, mask, score);
    k_softmax<<<B_, S_, 0, stream>>>(score);
    k_ctx<<<(B_ * E_) / 256, 256, 0, stream>>>(score, fbt, ctxbf);
    // concat projection as two accumulating NT GEMMs (no concat materialized)
    gemm(hbf + (size_t)(L_ - 1) * B_ * E_, E_, conc_bf, 2 * E_, cat, E_, E_, E_, 0);
    gemm(ctxbf, E_, conc_bf + E_, 2 * E_, cat, E_, E_, E_, 1);
    k_ln_tanh<<<B_, 256, 0, stream>>>(cat, ln_g, ln_b, catbf);
    // vocab projection -> logits for step t
    gemm(catbf, E_, proj_bf, E_, out + (size_t)t * B_ * V_, V_, V_, E_, 0);
  }
}
